// Transformer_79199196938374
// MI455X (gfx1250) — matrix operand order repaired and checked by reading, not compiled, not run on hardware
//
#include <hip/hip_runtime.h>
#include <hip/hip_bf16.h>
#include <math.h>

// ---------------- types for WMMA ----------------
typedef _Float16 half_t;
typedef __attribute__((ext_vector_type(16))) _Float16 v16h;
typedef __attribute__((ext_vector_type(8)))  _Float16 v8h;
typedef __attribute__((ext_vector_type(8)))  float    v8f;

#define N_NODES 1536
#define C_B   16
#define C_S   96
#define C_D   512
#define C_H   8
#define C_DK  64
#define C_DFF 2048
#define C_V   16384
#define C_NL  2

// ---- gfx1250 async-to-LDS path (ASYNCcnt) with safe fallback ----
#if __has_builtin(__builtin_amdgcn_global_load_async_to_lds_b128)
#define HAVE_ASYNC_LDS 1
// builtin expects: (int4 addrspace(1)* src, int4 addrspace(3)* dst, imm offset, imm cpol)
typedef int async_v4i __attribute__((__vector_size__(16)));
typedef __attribute__((address_space(1))) async_v4i* async_gptr;
typedef __attribute__((address_space(3))) async_v4i* async_lptr;
#else
#define HAVE_ASYNC_LDS 0
#endif

#if __has_builtin(__builtin_amdgcn_s_wait_asynccnt)
#define WAIT_ASYNC(n) __builtin_amdgcn_s_wait_asynccnt(n)
#else
#define WAIT_ASYNC(n) asm volatile("s_wait_asynccnt %0" :: "i"(n) : "memory")
#endif

// ---------------------------------------------------------------------------
// Embedding + positional encoding:  out[row, d] = emb[ids[row], d]*sqrt(D) + pe
// ---------------------------------------------------------------------------
__global__ __launch_bounds__(256)
void embed_kernel(const int* __restrict__ ids, const float* __restrict__ emb,
                  float* __restrict__ out, int nrows, int D, int S, float scale)
{
    size_t idx = (size_t)blockIdx.x * blockDim.x + threadIdx.x;
    if (idx >= (size_t)nrows * D) return;
    int row = (int)(idx / D);
    int d   = (int)(idx % D);
    int pos = row % S;
    int tok = ids[row];
    float div = __expf(-9.210340371976184f * (float)(d & ~1) / (float)D);
    float ang = (float)pos * div;
    float pe  = (d & 1) ? __cosf(ang) : __sinf(ang);
    out[idx] = emb[(size_t)tok * D + d] * scale + pe;
}

// ---------------------------------------------------------------------------
// LayerNorm (wave per row) -> f16 output (next GEMM's A operand)
// ---------------------------------------------------------------------------
__global__ __launch_bounds__(256)
void ln_f16_kernel(const float* __restrict__ x, const float* __restrict__ g,
                   const float* __restrict__ b, half_t* __restrict__ out,
                   int nrows, int D)
{
    int row  = blockIdx.x * blockDim.y + threadIdx.y;
    if (row >= nrows) return;
    int lane = threadIdx.x;
    const float* xr = x + (size_t)row * D;
    float s = 0.f, s2 = 0.f;
    for (int i = lane; i < D; i += 32) { float v = xr[i]; s += v; s2 += v * v; }
    #pragma unroll
    for (int off = 16; off; off >>= 1) {
        s  += __shfl_xor(s,  off, 32);
        s2 += __shfl_xor(s2, off, 32);
    }
    float m  = s / (float)D;
    float va = s2 / (float)D - m * m;
    float rs = rsqrtf(va + 1e-5f);
    half_t* orow = out + (size_t)row * D;
    for (int i = lane; i < D; i += 32)
        orow[i] = (half_t)(g[i] * (xr[i] - m) * rs + b[i]);
}

// ---------------------------------------------------------------------------
// fp32 -> f16 elementwise convert
// ---------------------------------------------------------------------------
__global__ __launch_bounds__(256)
void cvt_f16_kernel(const float* __restrict__ in, half_t* __restrict__ out, size_t n)
{
    size_t i = (size_t)blockIdx.x * blockDim.x + threadIdx.x;
    if (i < n) out[i] = (half_t)in[i];
}

// ---------------------------------------------------------------------------
// fp32 [K,N] -> f16 [N,K] transpose (weights: makes B fragments contiguous)
// ---------------------------------------------------------------------------
__global__ __launch_bounds__(256)
void cvtT_kernel(const float* __restrict__ in, half_t* __restrict__ out, int K, int N)
{
    size_t idx = (size_t)blockIdx.x * blockDim.x + threadIdx.x;
    if (idx >= (size_t)K * N) return;
    int k = (int)(idx / N);
    int n = (int)(idx % N);
    out[(size_t)n * K + k] = (half_t)in[idx];
}

// ---------------------------------------------------------------------------
// WMMA GEMM:  C[M,N] = A[M,K](f16) x BT[N,K](f16)^T  (+bias)(+res)(relu?)
// Block = 4 waves covering 64 M-rows x 64 N-cols. The 64x32 B tile is shared
// by all 4 waves: staged in LDS once per block via async-to-LDS (double
// buffered, ASYNCcnt), then consumed with ds_load + v_wmma_f32_16x16x32_f16.
// Requires M % 64 == 0, N % 64 == 0, K % 32 == 0 (all call sites satisfy).
//
// A-frag (16-bit A 16x32): lane m=lane&15, khalf=lane>>4,
//   a[0..7]=A[m][k0+khalf*8 .. +7], a[8..15]=A[m][k0+16+khalf*8 .. +7]
// B-frag (16-bit B 32x16): lane n=lane&15, khalf=lane>>4,
//   b[0..7]=BT[n][k0+khalf*8 .. +7], b[8..15]=BT[n][k0+16+khalf*8 .. +7]  -> LDS layout sB[buf][col*32 + kk]
// ---------------------------------------------------------------------------
__global__ __launch_bounds__(128)
void gemm_wmma_kernel(const half_t* __restrict__ A, const half_t* __restrict__ BT,
                      const float* __restrict__ bias, const float* __restrict__ res,
                      float* __restrict__ out32, half_t* __restrict__ out16,
                      int M, int N, int K, int relu)
{
    __shared__ _Float16 sB[2][64 * 32];            // 2 x 4 KB double buffer

    const int lane  = threadIdx.x;
    const int wv    = threadIdx.y;                 // wave id 0..3 -> M sub-tile
    const int tid   = wv * 32 + lane;              // 0..127
    const int m0    = (blockIdx.y * 4 + wv) * 16;
    const int n0    = blockIdx.x * 64;
    const int mlane = lane & 15;
    const int khalf = lane >> 4;

    const half_t* arow = A + (size_t)(m0 + mlane) * K + khalf * 8;

    // Stage the 64(cols) x 32(K) B tile for this k0 into LDS buffer p.
    // 256 x 16B chunks; 128 threads -> 2 async b128 issues each.
    auto loadB = [&](int p, int k0) {
        #pragma unroll
        for (int c = tid; c < 256; c += 128) {
            int col = c >> 2;
            int q   = c & 3;
            const half_t* g = BT + (size_t)(n0 + col) * K + k0 + q * 8;
            _Float16*     l = &sB[p][col * 32 + q * 8];
#if HAVE_ASYNC_LDS
            __builtin_amdgcn_global_load_async_to_lds_b128(
                (async_gptr)const_cast<half_t*>(g),
                (async_lptr)l,
                /*offset=*/0, /*cpol=*/0);
#else
            *(v8h*)l = *(const v8h*)g;
#endif
        }
    };

    v8f c[4];
    #pragma unroll
    for (int t = 0; t < 4; ++t) c[t] = (v8f){0.f,0.f,0.f,0.f,0.f,0.f,0.f,0.f};

    loadB(0, 0);                                   // prologue fill
    for (int k0 = 0; k0 < K; k0 += 32) {
        const int p = (k0 >> 5) & 1;
        const bool more = (k0 + 32) < K;
        if (more) loadB(p ^ 1, k0 + 32);           // prefetch next tile
#if HAVE_ASYNC_LDS
        if (more) WAIT_ASYNC(2);                   // older group landed; newer (2) in flight
        else      WAIT_ASYNC(0);
#endif
        __syncthreads();                           // tile p visible to all waves

        // A fragment (per-wave, direct from global)
        v16h a;
        {
            v8h lo = *(const v8h*)(arow + k0);
            v8h hi = *(const v8h*)(arow + k0 + 16);
            #pragma unroll
            for (int i = 0; i < 8; ++i) { a[i] = lo[i]; a[8 + i] = hi[i]; }
        }
        __builtin_prefetch(arow + k0 + 64, 0, 0);  // global_prefetch_b8

        #pragma unroll
        for (int t = 0; t < 4; ++t) {
            const _Float16* bl = &sB[p][(t * 16 + mlane) * 32 + khalf * 8];
            v16h b;
            {
                v8h lo = *(const v8h*)(bl);
                v8h hi = *(const v8h*)(bl + 16);
                #pragma unroll
                for (int i = 0; i < 8; ++i) { b[i] = lo[i]; b[8 + i] = hi[i]; }
            }
            c[t] = __builtin_amdgcn_wmma_f32_16x16x32_f16(
                       /*neg_a=*/false, a, /*neg_b=*/false, b,
                       /*c_mod=*/(short)0, c[t],
                       /*reuse_a=*/false, /*reuse_b=*/false);
        }
        __syncthreads();                           // done reading before overwrite
    }

    // C/D layout: VGPR r: lanes 0-15 -> (M=r, N=lane); lanes 16-31 -> (M=8+r, N=lane-16)
    #pragma unroll
    for (int t = 0; t < 4; ++t) {
        int col = n0 + t * 16 + mlane;
        float bv = bias ? bias[col] : 0.f;
        #pragma unroll
        for (int r = 0; r < 8; ++r) {
            int row = m0 + r + 8 * khalf;
            float val = c[t][r] + bv;
            if (res)  val += res[(size_t)row * N + col];
            if (relu) val  = fmaxf(val, 0.f);
            if (out32) out32[(size_t)row * N + col] = val;
            if (out16) out16[(size_t)row * N + col] = (half_t)val;
        }
    }
}

// ---------------------------------------------------------------------------
// Dense per-sequence attention, unnormalized clamped-exp scores (matches
// _attend over the full / causal / bipartite edge sets).
// Wave per query; lane owns dk-channels {lane, lane+32}.
// q,k,v,out: [B*S, H*DK] fp32
// ---------------------------------------------------------------------------
__global__ __launch_bounds__(256)
void attn_kernel(const float* __restrict__ q, const float* __restrict__ k,
                 const float* __restrict__ v, float* __restrict__ out,
                 int Sq, int H, int DKh, int causal)
{
    int lane = threadIdx.x;
    int qi   = blockIdx.y * blockDim.y + threadIdx.y;
    if (qi >= Sq) return;
    int b = blockIdx.x / H;
    int h = blockIdx.x % H;
    int Dm = H * DKh;

    size_t qbase = (size_t)(b * Sq + qi) * Dm + h * DKh;
    float q0 = q[qbase + lane];
    float q1 = q[qbase + lane + 32];

    float a0 = 0.f, a1 = 0.f, z = 0.f;
    int kend = causal ? (qi + 1) : Sq;
    for (int kk = 0; kk < kend; ++kk) {
        size_t kb = (size_t)(b * Sq + kk) * Dm + h * DKh;
        float d = q0 * k[kb + lane] + q1 * k[kb + lane + 32];
        #pragma unroll
        for (int off = 16; off; off >>= 1) d += __shfl_xor(d, off, 32);
        float s = __expf(fminf(fmaxf(d * 0.125f, -5.f), 5.f)); // 1/sqrt(64)
        a0 += s * v[kb + lane];
        a1 += s * v[kb + lane + 32];
        z  += s;
    }
    out[qbase + lane]      = a0 / z;
    out[qbase + lane + 32] = a1 / z;
}

// ---------------------------------------------------------------------------
// In-place log_softmax over rows of V
// ---------------------------------------------------------------------------
__global__ __launch_bounds__(256)
void logsoftmax_kernel(float* __restrict__ out, int V)
{
    __shared__ float red[256];
    float* r = out + (size_t)blockIdx.x * V;
    int tid = threadIdx.x;

    float mx = -1e30f;
    for (int i = tid; i < V; i += 256) mx = fmaxf(mx, r[i]);
    red[tid] = mx; __syncthreads();
    for (int s = 128; s; s >>= 1) { if (tid < s) red[tid] = fmaxf(red[tid], red[tid + s]); __syncthreads(); }
    mx = red[0]; __syncthreads();

    float sum = 0.f;
    for (int i = tid; i < V; i += 256) sum += __expf(r[i] - mx);
    red[tid] = sum; __syncthreads();
    for (int s = 128; s; s >>= 1) { if (tid < s) red[tid] += red[tid + s]; __syncthreads(); }
    float lse = mx + __logf(red[0]);
    __syncthreads();

    for (int i = tid; i < V; i += 256) r[i] -= lse;
}

// ---------------------------------------------------------------------------
// Host orchestration
// ---------------------------------------------------------------------------
extern "C" void kernel_launch(void* const* d_in, const int* in_sizes, int n_in,
                              void* d_out, int out_size, void* d_ws, size_t ws_size,
                              hipStream_t stream)
{
    const int B = C_B, S = C_S, D = C_D, H = C_H, DK = C_DK;
    const int DFF = C_DFF, V = C_V, NL = C_NL;
    const int N = N_NODES;
    (void)in_sizes; (void)n_in; (void)out_size; (void)ws_size;

    const int*   src_ids  = (const int*)  d_in[0];
    const int*   tgt_ids  = (const int*)  d_in[1];
    // d_in[2..7]: edge lists -> realized as dense full/causal attention
    const float* emb_src  = (const float*)d_in[8];
    const float* emb_tgt  = (const float*)d_in[9];
    const float* enc_w    = (const float*)d_in[10];
    const float* enc_ln_g = (const float*)d_in[11];
    const float* enc_ln_b = (const float*)d_in[12];
    const float* enc_w1   = (const float*)d_in[13];
    const float* enc_b1   = (const float*)d_in[14];
    const float* enc_w2   = (const float*)d_in[15];
    const float* enc_b2   = (const float*)d_in[16];
    const float* dec_w    = (const float*)d_in[17];
    const float* dec_ln_g = (const float*)d_in[18];
    const float* dec_ln_b = (const float*)d_in[19];
    const float* dec_w1   = (const float*)d_in[20];
    const float* dec_b1   = (const float*)d_in[21];
    const float* dec_w2   = (const float*)d_in[22];
    const float* dec_b2   = (const float*)d_in[23];
    const float* gen_w    = (const float*)d_in[24];
    const float* gen_b    = (const float*)d_in[25];

    // ---- workspace carve-up (256B aligned) ----
    char* wsp = (char*)d_ws;
    auto alloc = [&](size_t bytes) -> void* {
        void* p = (void*)wsp;
        wsp += (bytes + 255) & ~(size_t)255;
        return p;
    };
    float*  x    = (float*)alloc((size_t)N * D * 4);
    float*  y    = (float*)alloc((size_t)N * D * 4);
    float*  qb   = (float*)alloc((size_t)N * D * 4);
    float*  kb   = (float*)alloc((size_t)N * D * 4);
    float*  vb   = (float*)alloc((size_t)N * D * 4);
    float*  tmp  = (float*)alloc((size_t)N * D * 4);
    half_t* lnA  = (half_t*)alloc((size_t)N * D * 2);
    half_t* lnB  = (half_t*)alloc((size_t)N * D * 2);
    half_t* a16  = (half_t*)alloc((size_t)N * D * 2);
    half_t* h16  = (half_t*)alloc((size_t)N * DFF * 2);
    half_t* wT[8];
    for (int j = 0; j < 8; ++j) wT[j] = (half_t*)alloc((size_t)D * D * 2);
    half_t* w1T  = (half_t*)alloc((size_t)D * DFF * 2);
    half_t* w2T  = (half_t*)alloc((size_t)DFF * D * 2);
    half_t* genT = (half_t*)alloc((size_t)V * D * 2);

    // ---- launch helpers ----
    auto gemm = [&](const half_t* A, const half_t* BT, const float* bias,
                    const float* res, float* o32, half_t* o16,
                    int M, int Nn, int K, int relu) {
        dim3 grid(Nn / 64, M / 64);
        dim3 block(32, 4);
        gemm_wmma_kernel<<<grid, block, 0, stream>>>(A, BT, bias, res, o32, o16,
                                                     M, Nn, K, relu);
    };
    auto cvtT = [&](const float* in, half_t* out, int K, int Nn) {
        size_t n = (size_t)K * Nn;
        cvtT_kernel<<<(unsigned)((n + 255) / 256), 256, 0, stream>>>(in, out, K, Nn);
    };
    auto cvt = [&](const float* in, half_t* out, size_t n) {
        cvt_f16_kernel<<<(unsigned)((n + 255) / 256), 256, 0, stream>>>(in, out, n);
    };
    auto ln = [&](const float* xin, const float* g, const float* b, half_t* out) {
        ln_f16_kernel<<<dim3((N + 7) / 8), dim3(32, 8), 0, stream>>>(xin, g, b, out, N, D);
    };
    auto attn = [&](const float* q, const float* k, const float* v, float* o, int causal) {
        attn_kernel<<<dim3(B * H, S / 8), dim3(32, 8), 0, stream>>>(q, k, v, o, S, H, DK, causal);
    };

    const float scale = sqrtf((float)D);
    const size_t ND = (size_t)N * D;

    // ---- embeddings + posenc ----
    embed_kernel<<<(unsigned)((ND + 255) / 256), 256, 0, stream>>>(src_ids, emb_src, x, N, D, S, scale);
    embed_kernel<<<(unsigned)((ND + 255) / 256), 256, 0, stream>>>(tgt_ids, emb_tgt, y, N, D, S, scale);

    // ---- encoder layers ----
    for (int i = 0; i < NL; ++i) {
        const float* W  = enc_w + (size_t)i * 4 * D * D;
        const float* g0 = enc_ln_g + (size_t)(i * 2 + 0) * D;
        const float* b0 = enc_ln_b + (size_t)(i * 2 + 0) * D;
        const float* g1 = enc_ln_g + (size_t)(i * 2 + 1) * D;
        const float* b1 = enc_ln_b + (size_t)(i * 2 + 1) * D;

        ln(x, g0, b0, lnA);
        for (int j = 0; j < 4; ++j) cvtT(W + (size_t)j * D * D, wT[j], D, D);
        gemm(lnA, wT[0], nullptr, nullptr, qb, nullptr, N, D, D, 0);
        gemm(lnA, wT[1], nullptr, nullptr, kb, nullptr, N, D, D, 0);
        gemm(lnA, wT[2], nullptr, nullptr, vb, nullptr, N, D, D, 0);
        attn(qb, kb, vb, tmp, /*causal=*/0);
        cvt(tmp, a16, ND);
        gemm(a16, wT[3], nullptr, x, x, nullptr, N, D, D, 0);       // O-proj + residual

        ln(x, g1, b1, lnA);
        cvtT(enc_w1 + (size_t)i * D * DFF, w1T, D, DFF);
        gemm(lnA, w1T, enc_b1 + (size_t)i * DFF, nullptr, nullptr, h16, N, DFF, D, 1);
        cvtT(enc_w2 + (size_t)i * DFF * D, w2T, DFF, D);
        gemm(h16, w2T, enc_b2 + (size_t)i * D, x, x, nullptr, N, D, DFF, 0);
    }

    // ---- decoder layers ----
    for (int i = 0; i < NL; ++i) {
        const float* W  = dec_w + (size_t)i * 8 * D * D;
        const float* g0 = dec_ln_g + (size_t)(i * 3 + 0) * D;
        const float* b0 = dec_ln_b + (size_t)(i * 3 + 0) * D;
        const float* g1 = dec_ln_g + (size_t)(i * 3 + 1) * D;
        const float* b1 = dec_ln_b + (size_t)(i * 3 + 1) * D;
        const float* g2 = dec_ln_g + (size_t)(i * 3 + 2) * D;
        const float* b2 = dec_ln_b + (size_t)(i * 3 + 2) * D;

        // masked self-attention
        ln(y, g0, b0, lnA);
        for (int j = 0; j < 4; ++j) cvtT(W + (size_t)j * D * D, wT[j], D, D);
        gemm(lnA, wT[0], nullptr, nullptr, qb, nullptr, N, D, D, 0);
        gemm(lnA, wT[1], nullptr, nullptr, kb, nullptr, N, D, D, 0);
        gemm(lnA, wT[2], nullptr, nullptr, vb, nullptr, N, D, D, 0);
        attn(qb, kb, vb, tmp, /*causal=*/1);
        cvt(tmp, a16, ND);
        gemm(a16, wT[3], nullptr, y, y, nullptr, N, D, D, 0);

        // cross-attention (q from decoder, k/v from encoder memory; same LN params)
        ln(y, g1, b1, lnA);
        ln(x, g1, b1, lnB);
        for (int j = 4; j < 8; ++j) cvtT(W + (size_t)j * D * D, wT[j], D, D);
        gemm(lnA, wT[4], nullptr, nullptr, qb, nullptr, N, D, D, 0);
        gemm(lnB, wT[5], nullptr, nullptr, kb, nullptr, N, D, D, 0);
        gemm(lnB, wT[6], nullptr, nullptr, vb, nullptr, N, D, D, 0);
        attn(qb, kb, vb, tmp, /*causal=*/0);
        cvt(tmp, a16, ND);
        gemm(a16, wT[7], nullptr, y, y, nullptr, N, D, D, 0);

        // FFN
        ln(y, g2, b2, lnA);
        cvtT(dec_w1 + (size_t)i * D * DFF, w1T, D, DFF);
        gemm(lnA, w1T, dec_b1 + (size_t)i * DFF, nullptr, nullptr, h16, N, DFF, D, 1);
        cvtT(dec_w2 + (size_t)i * DFF * D, w2T, DFF, D);
        gemm(h16, w2T, dec_b2 + (size_t)i * D, y, y, nullptr, N, D, DFF, 0);
    }

    // ---- generator: logits = y @ gen_w + gen_b ; log_softmax rows ----
    cvt(y, a16, ND);
    cvtT(gen_w, genT, D, V);
    gemm(a16, genT, gen_b, nullptr, (float*)d_out, nullptr, N, V, D, 0);
    logsoftmax_kernel<<<N, 256, 0, stream>>>((float*)d_out, V);
}
